// BitConvFeatureExtractor_41652592836889
// MI455X (gfx1250) — matrix operand order repaired and checked by reading, not compiled, not run on hardware
//
#include <hip/hip_runtime.h>
#include <math.h>

typedef __attribute__((ext_vector_type(16))) _Float16 v16h;
typedef __attribute__((ext_vector_type(8)))  _Float16 v8h;
typedef __attribute__((ext_vector_type(8)))  float    v8f;

__device__ __forceinline__ float gelu_exact(float x) {
    return 0.5f * x * (1.0f + erff(x * 0.70710678118654752f));
}

// ---------------------------------------------------------------------------
// Zero-fill (halo zero page; ws is poisoned by the harness, so fill each call)
// ---------------------------------------------------------------------------
__global__ void zfill_kernel(_Float16* __restrict__ p, int count) {
    int i = blockIdx.x * blockDim.x + threadIdx.x;
    if (i < count) p[i] = (_Float16)0.0f;
}

// ---------------------------------------------------------------------------
// Layer-0 weight quantization: scale = max(mean|w|, 1e-5); q = clip(rint(w/s),-1,1)*s
// ---------------------------------------------------------------------------
__global__ void quant_l0_kernel(const float* __restrict__ w, float* __restrict__ wq, int count) {
    __shared__ float red[32];
    float s = 0.0f;
    for (int i = threadIdx.x; i < count; i += blockDim.x) s += fabsf(w[i]);
    for (int off = 16; off > 0; off >>= 1) s += __shfl_down(s, off, 32);
    int lane = threadIdx.x & 31, wid = threadIdx.x >> 5;
    if (lane == 0) red[wid] = s;
    __syncthreads();
    int nw = blockDim.x >> 5;
    if (wid == 0) {
        float v = (lane < nw) ? red[lane] : 0.0f;
        for (int off = 16; off > 0; off >>= 1) v += __shfl_down(v, off, 32);
        if (lane == 0) red[0] = v;
    }
    __syncthreads();
    float scale = fmaxf(red[0] / (float)count, 1e-5f);
    for (int i = threadIdx.x; i < count; i += blockDim.x) {
        float q = fminf(fmaxf(rintf(w[i] / scale), -1.0f), 1.0f) * scale;
        wq[i] = q;
    }
}

// ---------------------------------------------------------------------------
// Layers 1..6: ternary-quantize + pack weights into the wave32 WMMA B-fragment
// layout for v_wmma_f32_16x16x32_f16:
//   bp[((coutTile*K32 + chunk)*32 + lane)*16 + e]
//   within a chunk: K = (e&7) + 8*(lane>>4) + 16*(e>>3), cout = coutTile*16 + (lane&15)
//   reduction order r = kk*Cin + cin  (channel-fastest; matches channel-last acts)
// ---------------------------------------------------------------------------
__global__ void quant_pack_kernel(const float* __restrict__ w, _Float16* __restrict__ bp,
                                  int Cin, int K, int Cout) {
    __shared__ float red[32];
    const int count = Cout * Cin * K;
    float s = 0.0f;
    for (int i = threadIdx.x; i < count; i += blockDim.x) s += fabsf(w[i]);
    for (int off = 16; off > 0; off >>= 1) s += __shfl_down(s, off, 32);
    int lane = threadIdx.x & 31, wid = threadIdx.x >> 5;
    if (lane == 0) red[wid] = s;
    __syncthreads();
    int nw = blockDim.x >> 5;
    if (wid == 0) {
        float v = (lane < nw) ? red[lane] : 0.0f;
        for (int off = 16; off > 0; off >>= 1) v += __shfl_down(v, off, 32);
        if (lane == 0) red[0] = v;
    }
    __syncthreads();
    const float scale = fmaxf(red[0] / (float)count, 1e-5f);

    const int K32   = (Cin * K + 31) >> 5;
    const int total = (Cout >> 4) * K32 * 512;
    for (int idx = threadIdx.x; idx < total; idx += blockDim.x) {
        int e   = idx & 15;
        int ln  = (idx >> 4) & 31;
        int cc  = (idx >> 9) % K32;
        int ct  = (idx >> 9) / K32;
        int kin = (e & 7) + 8 * (ln >> 4) + 16 * (e >> 3);  // K index within 32-chunk
        int r   = cc * 32 + kin;            // global reduction index
        int co  = ct * 16 + (ln & 15);      // output channel
        _Float16 val = (_Float16)0.0f;
        if (r < Cin * K) {
            int cin = r % Cin;
            int kk  = r / Cin;
            float wv = w[((size_t)co * Cin + cin) * K + kk];
            float q  = fminf(fmaxf(rintf(wv / scale), -1.0f), 1.0f) * scale;
            val = (_Float16)q;
        }
        bp[idx] = val;
    }
}

// ---------------------------------------------------------------------------
// Layer 0: cin=1, k=10, s=5, p=4 — reduction dim 10, WMMA would waste 2/3 of K
// and it's 1.3% of the FLOPs. Direct kernel; writes channel-last f16 [N][32000][128].
// ---------------------------------------------------------------------------
__global__ void conv0_kernel(const float* __restrict__ x, const float* __restrict__ wq,
                             const float* __restrict__ bias, _Float16* __restrict__ out) {
    const size_t tid = (size_t)blockIdx.x * blockDim.x + threadIdx.x;
    const int cout = (int)(tid & 127);
    const int pos  = (int)((tid >> 7) % 32000);
    const int n    = (int)(tid / (128ull * 32000ull));
    if (n >= 8) return;
    const float* xr = x + (size_t)n * 160000;
    float acc = bias[cout];
    const int base = pos * 5 - 4;
#pragma unroll
    for (int kk = 0; kk < 10; ++kk) {
        int pin = base + kk;
        if (pin >= 0 && pin < 160000) acc += xr[pin] * wq[cout * 10 + kk];
    }
    out[tid] = (_Float16)gelu_exact(acc);
}

// ---------------------------------------------------------------------------
// WMMA implicit-GEMM conv, layers 1..6. Templated on CPK = Cin/32 and KTAPS = k
// so the whole reduction unrolls (no divisions; pure load + wmma).
//
// Register tile per wave: 4 pos-tiles x 2 cout-tiles (64 x 32 outputs,
// 8 f32 accumulators = 64 VGPRs). Each A fragment feeds 2 WMMAs, each B
// fragment feeds 4 WMMAs -> 12 b128 loads per 8 WMMAs (~0.75 KB/wave/WMMA),
// keeping the L0 load path under the matrix-unit issue rate.
// Block = 2 waves = 64 cout; packed B panels (~1.9 MB total) live in L2 (192MB).
//
// Fragments (wave32, v_wmma_f32_16x16x32_f16):
//   A (16x32): lane row M = lane&15; elems 0..7  -> K = 8*(lane>>4)+0..7
//                                    elems 8..15 -> K = 16+8*(lane>>4)+0..7
//   B (32x16): prepacked, 32 contiguous bytes per lane
//   D (16x16): vgpr v -> pos_local = v + 8*(lane>>4); col (cout) = lane&15
//
// Halo: per-tap 64-bit pointer select to a zeroed scratch page -> the unrolled
// chunk loop has no divergence (WMMA requires EXEC all-1s anyway).
// ---------------------------------------------------------------------------
template<int CPK, int KTAPS>
__launch_bounds__(64)
__global__ void conv_wmma_kernel(const _Float16* __restrict__ in,
                                 const _Float16* __restrict__ bp,
                                 const float* __restrict__ bias,
                                 _Float16* __restrict__ out16,
                                 float* __restrict__ out32,
                                 const _Float16* __restrict__ zpad,
                                 int Lin, int Lout, int Cout,
                                 int stride, int pad) {
    constexpr int Cin = CPK * 32;
    constexpr int K32 = CPK * KTAPS;

    const int lane = threadIdx.x & 31;
    const int wave = threadIdx.x >> 5;
    const int m    = lane & 15;
    const int hi   = lane >> 4;

    const int coutTile = blockIdx.y * 4 + wave * 2;  // this wave: tiles +0, +1
    const int posBase  = blockIdx.x * 64;
    const int n        = blockIdx.z;

    v8f acc[4][2] = {{{}, {}}, {{}, {}}, {{}, {}}, {{}, {}}};
    const _Float16* bpbase = bp + ((size_t)coutTile * K32 * 32 + lane) * 16;

#pragma unroll
    for (int kk = 0; kk < KTAPS; ++kk) {
        // Per-tap row pointers (hoisted out of the chunk loop); OOB -> zero page.
        const _Float16* ap[4];
#pragma unroll
        for (int t = 0; t < 4; ++t) {
            const int p   = posBase + t * 16 + m;
            const int pin = p * stride - pad + kk;
            const bool ok = (pin >= 0) && (pin < Lin);
            ap[t] = ok ? (in + ((size_t)n * Lin + pin) * Cin + hi * 8)
                       : (zpad + hi * 8);
        }
#pragma unroll
        for (int cc = 0; cc < CPK; ++cc) {
            const int c = kk * CPK + cc;
            const v16h b0 = *(const v16h*)(bpbase + (size_t)c * 512);
            const v16h b1 = *(const v16h*)(bpbase + (size_t)(K32 * 32 * 16) + (size_t)c * 512);
#pragma unroll
            for (int t = 0; t < 4; ++t) {
                union { v16h v; v8h h[2]; } a;
                a.h[0] = *(const v8h*)(ap[t] + cc * 32);       // K = hi*8 .. +7
                a.h[1] = *(const v8h*)(ap[t] + cc * 32 + 16);  // K = 16+hi*8 .. +7
                acc[t][0] = __builtin_amdgcn_wmma_f32_16x16x32_f16(
                    false, a.v, false, b0, (short)0, acc[t][0], false, false);
                acc[t][1] = __builtin_amdgcn_wmma_f32_16x16x32_f16(
                    false, a.v, false, b1, (short)0, acc[t][1], false, false);
            }
        }
    }

#pragma unroll
    for (int ct = 0; ct < 2; ++ct) {
        const int   cout = (coutTile + ct) * 16 + m;
        const float bb   = bias[cout];
#pragma unroll
        for (int t = 0; t < 4; ++t) {
#pragma unroll
            for (int v = 0; v < 8; ++v) {
                const int pos = posBase + t * 16 + v + 8 * hi;
                if (pos < Lout) {
                    const float g = gelu_exact(acc[t][ct][v] + bb);
                    out16[((size_t)n * Lout + pos) * Cout + cout] = (_Float16)g;
                    if (out32) out32[((size_t)n * Cout + cout) * Lout + pos] = g;
                }
            }
        }
    }
}

// ---------------------------------------------------------------------------
// Host launcher
// ---------------------------------------------------------------------------
extern "C" void kernel_launch(void* const* d_in, const int* in_sizes, int n_in,
                              void* d_out, int out_size, void* d_ws, size_t ws_size,
                              hipStream_t stream) {
    (void)in_sizes; (void)n_in; (void)out_size; (void)ws_size;

    const float* x = (const float*)d_in[0];
    const float* w[7]; const float* b[7];
    for (int i = 0; i < 7; ++i) { w[i] = (const float*)d_in[1 + 2 * i]; b[i] = (const float*)d_in[2 + 2 * i]; }

    struct LayerDef { int cin, cout, k, s, p, lin, lout; };
    static const LayerDef Ls[7] = {
        {  1, 128, 10, 5, 4, 160000, 32000},
        {128, 192,  3, 2, 1,  32000, 16000},
        {192, 192,  3, 2, 1,  16000,  8000},
        {192, 192,  3, 2, 1,   8000,  4000},
        {192, 256,  3, 2, 1,   4000,  2000},
        {256, 256,  4, 2, 1,   2000,  1000},
        {256, 256,  4, 2, 1,   1000,   500},
    };

    char* ws = (char*)d_ws;
    size_t off = 0;
    auto balloc = [&](size_t bytes) -> char* {
        char* p = ws + off;
        off += (bytes + 255) & ~(size_t)255;
        return p;
    };

    _Float16* actA = (_Float16*)balloc(8ull * 32000 * 128 * sizeof(_Float16)); // max even-layer act
    _Float16* actB = (_Float16*)balloc(8ull * 16000 * 192 * sizeof(_Float16)); // max odd-layer act
    float*    wq0  = (float*)   balloc(128 * 10 * sizeof(float));
    _Float16* zpad = (_Float16*)balloc(512 * sizeof(_Float16));                // halo zero page
    _Float16* bpack[7] = {};
    for (int i = 1; i < 7; ++i) {
        const int K32 = (Ls[i].cin * Ls[i].k + 31) >> 5;
        bpack[i] = (_Float16*)balloc((size_t)(Ls[i].cout >> 4) * K32 * 512 * sizeof(_Float16));
    }

    // Setup: zero page + weight quantization/packing (tiny; off critical path)
    zfill_kernel<<<2, 256, 0, stream>>>(zpad, 512);
    quant_l0_kernel<<<1, 256, 0, stream>>>(w[0], wq0, 128 * 10);
    for (int i = 1; i < 7; ++i)
        quant_pack_kernel<<<1, 1024, 0, stream>>>(w[i], bpack[i], Ls[i].cin, Ls[i].k, Ls[i].cout);

    // Layer 0: direct conv (cin*k = 10)
    conv0_kernel<<<(8 * 32000 * 128) / 256, 256, 0, stream>>>(x, wq0, b[0], actA);

    // Layers 1..6: WMMA implicit GEMM, ping-pong activation buffers
    _Float16* cur = actA;
    _Float16* nxt = actB;
    for (int i = 1; i < 7; ++i) {
        const LayerDef& L = Ls[i];
        dim3 grid((L.lout + 63) / 64, L.cout / 64, 8);
        float* o32 = (i == 6) ? (float*)d_out : nullptr;
        if (L.cin == 128)
            conv_wmma_kernel<4, 3><<<grid, 64, 0, stream>>>(cur, bpack[i], b[i], nxt, o32, zpad,
                                                            L.lin, L.lout, L.cout, L.s, L.p);
        else if (L.cin == 192)
            conv_wmma_kernel<6, 3><<<grid, 64, 0, stream>>>(cur, bpack[i], b[i], nxt, o32, zpad,
                                                            L.lin, L.lout, L.cout, L.s, L.p);
        else
            conv_wmma_kernel<8, 4><<<grid, 64, 0, stream>>>(cur, bpack[i], b[i], nxt, o32, zpad,
                                                            L.lin, L.lout, L.cout, L.s, L.p);
        _Float16* t = cur; cur = nxt; nxt = t;
    }
}
